// Self_Attention_27522150433268
// MI455X (gfx1250) — matrix operand order repaired and checked by reading, not compiled, not run on hardware
//
#include <hip/hip_runtime.h>

// ---------------------------------------------------------------------------
// Self-attention (B=16, C=256, H=W=64) for gfx1250 — wave32 + WMMA f16->f32,
// LDS-staged GEMM operands, TDM (tensor_load_to_lds) V staging.
// Two-pass column-softmax attention (softmax over the QUERY axis) that never
// materializes the 256MB attention matrix.
// ---------------------------------------------------------------------------

typedef __attribute__((ext_vector_type(16))) _Float16 v16h;
typedef __attribute__((ext_vector_type(8)))  _Float16 v8h;
typedef __attribute__((ext_vector_type(8)))  float    v8f;
typedef __attribute__((ext_vector_type(4)))  unsigned int v4u;
typedef __attribute__((ext_vector_type(4)))  int      v4i;
typedef __attribute__((ext_vector_type(8)))  int      v8i;

#define NB   16
#define NC   256
#define HW   4096      // 64*64
#define DN   1024      // 32*32 after 2x2 maxpool
#define C8   32
#define C2   128

// A-operand (16x32 f16): lane l -> row m = l&15, half h = l>>4.
// elements 0..15 = K(k0 + h*16 + e).
__device__ __forceinline__ v16h load_a16(const _Float16* row, int h, int k0) {
  return *(const v16h*)(row + k0 + h * 16);
}

__device__ __forceinline__ v8f wmma_f16(v16h a, v16h b, v8f c) {
  return __builtin_amdgcn_wmma_f32_16x16x32_f16(false, a, false, b, (short)0, c,
                                                false, false);
}

// Issue a 1-D TDM copy: `bytes/2` f16 elements from global -> LDS (async,
// tracked by TENSORcnt). Tile is contiguous: tile_dim0 = nelem, dims 1+ unused.
__device__ __forceinline__ void tdm_load_1d(const void* gsrc, unsigned lds_off,
                                            unsigned nelem) {
  unsigned long long ga = (unsigned long long)(size_t)gsrc;
  v4u g0 = { 1u,                                   // count=1 (valid descriptor)
             lds_off,                              // LDS byte address
             (unsigned)(ga & 0xffffffffu),
             (unsigned)((ga >> 32) & 0x01ffffffu) | (2u << 30) };  // type=2
  const unsigned td0 = 1u << 20;                   // tensor_dim0 (OOB bound)
  v8i g1 = { (int)(1u << 16),                      // data_size=1 -> 2 bytes
             (int)((td0 & 0xffffu) << 16),         // tensor_dim0[15:0]
             (int)((td0 >> 16) | (1u << 16)),      // tensor_dim0[31:16] | tdim1=1
             (int)(nelem << 16),                   // tensor_dim1[31:16]=0 | tile_dim0
             0, 0, 0, 0 };                         // tile_dim1/2=0, strides=0
  v4i gz = { 0, 0, 0, 0 };
#if defined(__clang_major__) && __clang_major__ >= 23
  v8i gz8 = { 0, 0, 0, 0, 0, 0, 0, 0 };
  __builtin_amdgcn_tensor_load_to_lds(g0, g1, gz, gz, gz8, 0);
#else
  __builtin_amdgcn_tensor_load_to_lds(g0, g1, gz, gz, 0);
#endif
}

// ---------------------------------------------------------------------------
__global__ void f32_to_f16(const float* __restrict__ src,
                           _Float16* __restrict__ dst, int n) {
  int i = blockIdx.x * blockDim.x + threadIdx.x;
  if (i < n) dst[i] = (_Float16)src[i];
}

// out[b][o][s] = sum_k W[o][k] * xh[b][k][s] + bias[o]  (natural layout, f16).
// One wave per 16-column block: stage the whole [NC][16] x-panel transposed in
// LDS with 32B vector loads, then loop all O/16 row tiles reusing the panel.
__global__ void conv1x1_wmma(const _Float16* __restrict__ xh,
                             const _Float16* __restrict__ wh,
                             const float* __restrict__ bias,
                             _Float16* __restrict__ outh, int O) {
  const int lane = threadIdx.x & 31;
  const int hv   = lane >> 4;
  const int n    = lane & 15;
  const int s0   = blockIdx.x * 16;
  const int b    = blockIdx.y;

  __shared__ __align__(64) _Float16 xT[16 * NC];   // [s][k] transposed panel

  const _Float16* xb = xh + (size_t)b * NC * HW;
  #pragma unroll
  for (int r = 0; r < NC / 32; ++r) {              // 8 coalesced 32B rows/lane
    int k = r * 32 + lane;
    v16h row = *(const v16h*)(xb + (size_t)k * HW + s0);
    #pragma unroll
    for (int e = 0; e < 16; ++e) xT[e * NC + k] = row[e];
  }
  __syncthreads();

  for (int ot = 0; ot < O / 16; ++ot) {
    const _Float16* wrow = wh + (size_t)(ot * 16 + n) * NC;  // A row (m=l&15)
    v8f acc = {};
    #pragma unroll
    for (int kk = 0; kk < NC; kk += 32) {
      v16h a  = load_a16(wrow, hv, kk);
      v16h bv = *(const v16h*)(xT + (size_t)n * NC + kk + hv * 16);
      acc = wmma_f16(a, bv, acc);
    }
    #pragma unroll
    for (int v = 0; v < 8; ++v) {
      int o = ot * 16 + 8 * hv + v;                // D: row = 8h+v, col = n
      outh[(size_t)b * O * HW + (size_t)o * HW + s0 + n] =
          (_Float16)(acc[v] + bias[o]);
    }
  }
}

// 2x2 max pool over the spatial 64x64 -> 32x32, natural [b][o][y][x] layout.
__global__ void maxpool2x2(const _Float16* __restrict__ in,
                           _Float16* __restrict__ out, int total, int O) {
  int i = blockIdx.x * blockDim.x + threadIdx.x;
  if (i >= total) return;
  int x = i & 31;
  int y = (i >> 5) & 31;
  int o = (i >> 10) % O;
  int b = (i >> 10) / O;
  const _Float16* p = in + (size_t)(b * O + o) * HW + (size_t)(2 * y) * 64 + 2 * x;
  float a00 = (float)p[0],  a01 = (float)p[1];
  float a10 = (float)p[64], a11 = (float)p[65];
  out[i] = (_Float16)fmaxf(fmaxf(a00, a01), fmaxf(a10, a11));
}

// Pass 1: per-column (k-axis) online max and sum of exp over all 4096 queries.
// S = Q(4096x32) * K^T(1024x32)^T ; both operands are contiguous 32-half rows.
__global__ void attn_stats(const _Float16* __restrict__ qh,
                           const _Float16* __restrict__ kth,
                           float* __restrict__ cmax, float* __restrict__ csum) {
  const int lane = threadIdx.x & 31;
  const int hv   = lane >> 4;
  const int n    = lane & 15;
  const int j0   = blockIdx.x * 16;
  const int b    = blockIdx.y;

  const _Float16* qb   = qh + (size_t)b * HW * C8;
  const _Float16* krow = kth + (size_t)b * DN * C8 + (size_t)(j0 + n) * C8;
  v16h bv = *(const v16h*)(krow + hv * 16);            // K tile fixed per wave

  float m = -3.0e38f, ssum = 0.0f;
  for (int it = 0; it < HW / 16; ++it) {
    const _Float16* qrow = qb + (size_t)(it * 16 + n) * C8;
    __builtin_prefetch(qrow + 16 * C8, 0, 0);          // global_prefetch_b8
    v16h a = load_a16(qrow, hv, 0);
    v8f  d = {};
    d = wmma_f16(a, bv, d);
    float mt = d[0];
    #pragma unroll
    for (int v = 1; v < 8; ++v) mt = fmaxf(mt, d[v]);
    if (mt > m) { ssum *= __expf(m - mt); m = mt; }
    #pragma unroll
    for (int v = 0; v < 8; ++v) ssum += __expf(d[v] - m);
  }
  // lane l covers rows {8h..8h+7} of column j0+n; combine the two halves.
  float mo = __shfl_xor(m, 16, 32);
  float so = __shfl_xor(ssum, 16, 32);
  float M  = fmaxf(m, mo);
  float S  = ssum * __expf(m - M) + so * __expf(mo - M);
  if (hv == 0) {
    cmax[b * DN + j0 + n] = M;
    csum[b * DN + j0 + n] = S;
  }
}

// Pass 2: recompute S tile-by-tile, form P = exp(S - cmax)/csum, and fuse
// applied(16x128) += P(16x32) * V(32x128). V tiles arrive via TDM into LDS
// (async, overlapped with the S-tile WMMAs), then get re-laid for B operands.
__global__ void attn_apply(const _Float16* __restrict__ qh,
                           const _Float16* __restrict__ kth,
                           const _Float16* __restrict__ vth,
                           const float* __restrict__ cmax,
                           const float* __restrict__ csum,
                           _Float16* __restrict__ apph) {
  const int lane = threadIdx.x & 31;
  const int hv   = lane >> 4;
  const int n    = lane & 15;
  const int i0   = blockIdx.x * 16;
  const int b    = blockIdx.y;

  __shared__ __align__(64)  _Float16 pT[16 * 32];    // P tile, row-major [i][j]
  __shared__ __align__(64)  _Float16 vT[C2 * 32];    // V^T stage, [c][j]
  __shared__ __align__(128) _Float16 vraw[32 * C2];  // TDM landing, [j][c]

  const _Float16* qrow = qh + (size_t)b * HW * C8 + (size_t)(i0 + n) * C8;
  v16h aQ = load_a16(qrow, hv, 0);                   // Q tile fixed per wave
  const _Float16* kb = kth + (size_t)b * DN * C8;
  const _Float16* vb = vth + (size_t)b * DN * C2;
  const unsigned vraw_off = (unsigned)(size_t)&vraw[0];

  v8f zero = {};
  v8f acc[8];
  #pragma unroll
  for (int t = 0; t < 8; ++t) acc[t] = zero;

  for (int jb = 0; jb < DN / 32; ++jb) {
    const int j0 = jb * 32;
    // async: DMA the contiguous 8KB V[j0..j0+31][0..127] tile into LDS
    tdm_load_1d(vb + (size_t)j0 * C2, vraw_off, 32 * C2);

    // meanwhile: two 16-col S tiles -> P stored to LDS in [16 x 32] row-major
    #pragma unroll
    for (int t = 0; t < 2; ++t) {
      int jc = j0 + t * 16 + n;
      v16h bv = *(const v16h*)(kb + (size_t)jc * C8 + hv * 16);
      v8f  d  = {};
      d = wmma_f16(aQ, bv, d);
      float mj  = cmax[b * DN + jc];
      float inv = 1.0f / csum[b * DN + jc];
      #pragma unroll
      for (int v = 0; v < 8; ++v)
        pT[(8 * hv + v) * 32 + t * 16 + n] = (_Float16)(__expf(d[v] - mj) * inv);
    }

    __builtin_amdgcn_s_wait_tensorcnt(0);            // TDM tile landed
    // transpose V tile in LDS: lane l owns row j0+l -> vT[c][l]
    {
      const _Float16* vrow = &vraw[lane * C2];
      #pragma unroll 8
      for (int c = 0; c < C2; ++c) vT[c * 32 + lane] = vrow[c];
    }
    __syncthreads();
    v16h aP = load_a16(&pT[n * 32], hv, 0);
    #pragma unroll
    for (int ct = 0; ct < 8; ++ct) {
      v16h bV = *(const v16h*)(vT + (size_t)(ct * 16 + n) * 32 + hv * 16);
      acc[ct] = wmma_f16(aP, bV, acc[ct]);
    }
    __syncthreads();
  }
  #pragma unroll
  for (int ct = 0; ct < 8; ++ct) {
    #pragma unroll
    for (int v = 0; v < 8; ++v)
      apph[(size_t)b * HW * C2 + (size_t)(i0 + 8 * hv + v) * C2 + ct * 16 + n] =
          (_Float16)acc[ct][v];
  }
}

// Final: out = gamma * (W2(256x128) @ applied^raw(128x4096) + b2) + x.
// Same LDS-panel scheme as conv1x1: stage [128][16] transposed once, loop all
// 16 output row tiles.
__global__ void final_conv(const _Float16* __restrict__ apph,
                           const _Float16* __restrict__ w2h,
                           const float* __restrict__ b2,
                           const float* __restrict__ gamma,
                           const float* __restrict__ x,
                           float* __restrict__ out) {
  const int lane = threadIdx.x & 31;
  const int hv   = lane >> 4;
  const int n    = lane & 15;
  const int s0   = blockIdx.x * 16;
  const int b    = blockIdx.y;

  __shared__ __align__(64) _Float16 aT[16 * C2];     // [s][ch] transposed panel

  const _Float16* ab = apph + (size_t)b * C2 * HW;   // raw view [128][4096]
  #pragma unroll
  for (int r = 0; r < C2 / 32; ++r) {                // 4 coalesced 32B rows/lane
    int ch = r * 32 + lane;
    v16h row = *(const v16h*)(ab + (size_t)ch * HW + s0);
    #pragma unroll
    for (int e = 0; e < 16; ++e) aT[e * C2 + ch] = row[e];
  }
  __syncthreads();

  float g = gamma[0];
  for (int ot = 0; ot < NC / 16; ++ot) {
    const _Float16* wrow = w2h + (size_t)(ot * 16 + n) * C2;
    v8f acc = {};
    #pragma unroll
    for (int kk = 0; kk < C2; kk += 32) {
      v16h a  = load_a16(wrow, hv, kk);
      v16h bv = *(const v16h*)(aT + (size_t)n * C2 + kk + hv * 16);
      acc = wmma_f16(a, bv, acc);
    }
    #pragma unroll
    for (int v = 0; v < 8; ++v) {
      int o = ot * 16 + 8 * hv + v;
      size_t idx = (size_t)b * NC * HW + (size_t)o * HW + s0 + n;
      out[idx] = g * (acc[v] + b2[o]) + x[idx];
    }
  }
}

// ---------------------------------------------------------------------------
extern "C" void kernel_launch(void* const* d_in, const int* in_sizes, int n_in,
                              void* d_out, int out_size, void* d_ws,
                              size_t ws_size, hipStream_t stream) {
  const float* x     = (const float*)d_in[0];
  const float* qw    = (const float*)d_in[1];
  const float* qb    = (const float*)d_in[2];
  const float* kw    = (const float*)d_in[3];
  const float* kbias = (const float*)d_in[4];
  const float* vw    = (const float*)d_in[5];
  const float* vbias = (const float*)d_in[6];
  const float* w2    = (const float*)d_in[7];
  const float* b2    = (const float*)d_in[8];
  const float* gamma = (const float*)d_in[9];

  char* ws = (char*)d_ws;
  size_t cur = 0;
  auto carve = [&](size_t bytes) -> char* {
    char* p = ws + cur;
    cur += (bytes + 255) & ~(size_t)255;
    return p;
  };

  const int N_X = NB * NC * HW;                 // 16,777,216
  _Float16* xh    = (_Float16*)carve((size_t)N_X * 2);
  _Float16* qwh   = (_Float16*)carve((size_t)C8 * NC * 2);
  _Float16* kwh   = (_Float16*)carve((size_t)C8 * NC * 2);
  _Float16* vwh   = (_Float16*)carve((size_t)C2 * NC * 2);
  _Float16* w2h   = (_Float16*)carve((size_t)NC * C2 * 2);
  _Float16* qhb   = (_Float16*)carve((size_t)NB * C8 * HW * 2);
  _Float16* kfull = (_Float16*)carve((size_t)NB * C8 * HW * 2);
  _Float16* vfull = (_Float16*)carve((size_t)NB * C2 * HW * 2);
  _Float16* kth   = (_Float16*)carve((size_t)NB * C8 * DN * 2);
  _Float16* vth   = (_Float16*)carve((size_t)NB * C2 * DN * 2);
  float*    cmax  = (float*)carve((size_t)NB * DN * 4);
  float*    csum  = (float*)carve((size_t)NB * DN * 4);
  _Float16* apph  = (_Float16*)carve((size_t)NB * HW * C2 * 2);

  // 1) f32 -> f16 conversions
  f32_to_f16<<<(N_X + 255) / 256, 256, 0, stream>>>(x, xh, N_X);
  f32_to_f16<<<(C8 * NC + 255) / 256, 256, 0, stream>>>(qw, qwh, C8 * NC);
  f32_to_f16<<<(C8 * NC + 255) / 256, 256, 0, stream>>>(kw, kwh, C8 * NC);
  f32_to_f16<<<(C2 * NC + 255) / 256, 256, 0, stream>>>(vw, vwh, C2 * NC);
  f32_to_f16<<<(NC * C2 + 255) / 256, 256, 0, stream>>>(w2, w2h, NC * C2);

  // 2) 1x1 convs (GEMMs, K=256) via WMMA with LDS-staged panels
  conv1x1_wmma<<<dim3(HW / 16, NB), 32, 0, stream>>>(xh, qwh, qb, qhb, C8);
  conv1x1_wmma<<<dim3(HW / 16, NB), 32, 0, stream>>>(xh, kwh, kbias, kfull, C8);
  conv1x1_wmma<<<dim3(HW / 16, NB), 32, 0, stream>>>(xh, vwh, vbias, vfull, C2);

  // 3) 2x2 maxpool for key / value paths
  maxpool2x2<<<(NB * C8 * DN + 255) / 256, 256, 0, stream>>>(kfull, kth, NB * C8 * DN, C8);
  maxpool2x2<<<(NB * C2 * DN + 255) / 256, 256, 0, stream>>>(vfull, vth, NB * C2 * DN, C2);

  // 4) pass 1: column (query-axis) softmax statistics
  attn_stats<<<dim3(DN / 16, NB), 32, 0, stream>>>(qhb, kth, cmax, csum);

  // 5) pass 2: recompute S, P = exp/norm, fused P @ V with TDM V staging
  attn_apply<<<dim3(HW / 16, NB), 32, 0, stream>>>(qhb, kth, vth, cmax, csum, apph);

  // 6) final 1x1 conv + gamma residual
  final_conv<<<dim3(HW / 16, NB), 32, 0, stream>>>(apph, w2h, b2, gamma, x,
                                                  (float*)d_out);
}
